// training_GCN_20220706029993
// MI455X (gfx1250) — compile-verified, matrix operand order checked by reading, not run on hardware
//
#include <hip/hip_runtime.h>
#include <stdint.h>

// ---------------------------------------------------------------------------
// MI455X (gfx1250) GCN-training forward pass.
//
// All GEMMs run on v_wmma_f32_16x16x32_bf16 (fp32 accumulate). Activations
// are materialized in bf16 by their producers (cvt kernels / fused epilogues)
// so the GEMM inner loop is pure b128 loads + WMMA:
//   A fragment (16x32 bf16): lane L -> M = L&15, half = L>>4; the lane's 16
//     K-values are the two contiguous byte-16 groups  [k0+8*half .. +7] and
//     [k0+16+8*half .. +7]  -> two b128 loads, stitched by shufflevector.
//   B fragment: identical striping with N = L&15; weights pre-transposed to
//     [F,K] bf16 so B loads are the same two b128 loads per n-tile.
//   C/D (16x16 f32): lane L -> N = L&15, VGPR r -> row M = r + 8*half.
//
// Roofline: GEMMs total ~29 GFLOP -> negligible on the WMMA pipe. The edge
// scatter (2x 800k edges x 128 feats + 1x 48 feats of gathers + f32 atomics,
// ~2 GB of L2 traffic) dominates at 23.3 TB/s; it is vectorized float4.
// ---------------------------------------------------------------------------

typedef __attribute__((ext_vector_type(16))) __bf16 v16bf;
typedef __attribute__((ext_vector_type(8)))  __bf16 bf16x8;
typedef __attribute__((ext_vector_type(4)))  __bf16 bf16x4;
typedef __attribute__((ext_vector_type(8)))  float  v8f;

#define IN_F   512
#define HID_F  128
#define OUT_F  40
#define OUT_P  48     // OUT padded to 3 WMMA n-tiles
#define CHID_F 256

__device__ __forceinline__ __bf16 f2bf(float f) {
  union { float f; uint32_t u; } v; v.f = f;
  uint32_t r = v.u + 0x7fffu + ((v.u >> 16) & 1u);   // round-to-nearest-even
  unsigned short h = (unsigned short)(r >> 16);
  __bf16 o;
  __builtin_memcpy(&o, &h, sizeof(h));
  return o;
}

// deterministic stateless dropout mask, p=0.5 keep, scale 2.0
__device__ __forceinline__ float dropmask(uint32_t idx, uint32_t seed) {
  uint32_t x = idx * 2654435761u + seed;
  x ^= x >> 16; x *= 0x7feb352du;
  x ^= x >> 15; x *= 0x846ca68bu;
  x ^= x >> 16;
  return (x & 1u) ? 2.0f : 0.0f;
}

__device__ __forceinline__ v16bf load_frag(const __bf16* base, int k0) {
  bf16x8 lo = *(const bf16x8*)(base + k0);
  bf16x8 hi = *(const bf16x8*)(base + k0 + 16);
  return __builtin_shufflevector(lo, hi, 0, 1, 2, 3, 4, 5, 6, 7,
                                 8, 9, 10, 11, 12, 13, 14, 15);
}

// ---------------------------------------------------------------------------
// C[N,F] = A[N,K](bf16) * Wt[F,K](bf16 transposed), fp32 accumulate.
// One wave per 16-row stripe, NT 16-col tiles per chunk.
// flags: 1=bias, 2=relu, 4=dropout, 8=store bf16 (to Cb) else f32 (to Cf)
// ---------------------------------------------------------------------------
template <int NT>
__global__ void gemm_bf16_wmma(const __bf16* __restrict__ A, int K,
                               const __bf16* __restrict__ Wt, int F,
                               const float* __restrict__ bias, int biasN,
                               float* __restrict__ Cf, __bf16* __restrict__ Cb,
                               int N, int flags, uint32_t seed) {
  const int lane = threadIdx.x & 31;
  const int wave = threadIdx.x >> 5;
  const int tile = blockIdx.x * (blockDim.x >> 5) + wave;
  const int row0 = tile * 16;
  if (row0 >= N) return;

  const int m    = lane & 15;
  const int half = lane >> 4;
  const __bf16* Ar = A + (size_t)(row0 + m) * K + 8 * half;

  for (int n0 = 0; n0 < F; n0 += NT * 16) {
    v8f acc[NT] = {};

    for (int k0 = 0; k0 < K; k0 += 32) {
      const v16bf a = load_frag(Ar, k0);
#pragma unroll
      for (int t = 0; t < NT; ++t) {
        const int n = n0 + t * 16 + m;
        const v16bf b = load_frag(Wt + (size_t)n * K + 8 * half, k0);
        acc[t] = __builtin_amdgcn_wmma_f32_16x16x32_bf16(
            false, a, false, b, (short)0, acc[t], false, false);
      }
    }

#pragma unroll
    for (int t = 0; t < NT; ++t) {
      const int n = n0 + t * 16 + m;
      float bv = 0.0f;
      if ((flags & 1) && n < biasN) bv = bias[n];
#pragma unroll
      for (int r = 0; r < 8; ++r) {
        const int row = row0 + r + 8 * half;
        float v = acc[t][r] + bv;
        if (flags & 2) v = fmaxf(v, 0.0f);
        if (flags & 4) v *= dropmask((uint32_t)(row * F + n), seed);
        if (flags & 8) Cb[(size_t)row * F + n] = f2bf(v);
        else           Cf[(size_t)row * F + n] = v;
      }
    }
  }
}

// --------------------------- elementwise kernels ----------------------------

__global__ void fill_one(float* __restrict__ p, int n) {
  int i = blockIdx.x * blockDim.x + threadIdx.x;
  if (i < n) p[i] = 1.0f;
}

__global__ void deg_edges(const int* __restrict__ dst, float* __restrict__ deg, int E) {
  int i = blockIdx.x * blockDim.x + threadIdx.x;
  if (i < E) atomicAdd(&deg[dst[i]], 1.0f);
}

__global__ void to_rsqrt(float* __restrict__ p, int n) {
  int i = blockIdx.x * blockDim.x + threadIdx.x;
  if (i < n) p[i] = rsqrtf(p[i]);
}

// f32 -> bf16, 4 elements per thread (n must be a multiple of 4)
__global__ void cvtbf(const float* __restrict__ in, __bf16* __restrict__ out,
                      long long n4) {
  long long i = (long long)blockIdx.x * blockDim.x + threadIdx.x;
  if (i >= n4) return;
  const float4 f = ((const float4*)in)[i];
  bf16x4 o;
  o[0] = f2bf(f.x); o[1] = f2bf(f.y); o[2] = f2bf(f.z); o[3] = f2bf(f.w);
  ((bf16x4*)out)[i] = o;
}

// W[K,F] f32 row-major -> Wt[Fp,K] bf16 (transposed, zero-padded cols F..Fp-1)
__global__ void wconv(const float* __restrict__ W, __bf16* __restrict__ Wt,
                      int K, int F, int Fp) {
  int i = blockIdx.x * blockDim.x + threadIdx.x;
  if (i >= Fp * K) return;
  int n = i / K, k = i - n * K;
  Wt[i] = (n < F) ? f2bf(W[(size_t)k * F + n]) : f2bf(0.0f);
}

// AGG[dst] += H[src] * dis[src]*dis[dst]; one thread per (edge, 4 features)
__global__ void edge_agg(const int* __restrict__ src, const int* __restrict__ dst,
                         const float* __restrict__ dis, const float* __restrict__ H,
                         float* __restrict__ AGG, int E, int F, int F4) {
  long long i = (long long)blockIdx.x * blockDim.x + threadIdx.x;
  if (i >= (long long)E * F4) return;
  const int e = (int)(i / F4);
  const int q = (int)(i % F4) * 4;
  const int s = src[e], d = dst[e];
  const float coef = dis[s] * dis[d];
  const float4 h = *(const float4*)(H + (size_t)s * F + q);
  float* a = AGG + (size_t)d * F + q;
  atomicAdd(a + 0, h.x * coef);
  atomicAdd(a + 1, h.y * coef);
  atomicAdd(a + 2, h.z * coef);
  atomicAdd(a + 3, h.w * coef);
}

// v = AGG + HW*dis^2 + bias (+relu/dropout); write f32 (outF) or bf16 (outB)
__global__ void gcn_finalize(const float* __restrict__ AGG, const float* __restrict__ HW,
                             const float* __restrict__ dis, const float* __restrict__ bias,
                             int N, int F, int biasN, int flags, uint32_t seed,
                             float* __restrict__ outF, __bf16* __restrict__ outB) {
  long long i = (long long)blockIdx.x * blockDim.x + threadIdx.x;
  if (i >= (long long)N * F) return;
  const int row = (int)(i / F), f = (int)(i % F);
  const float d = dis[row];
  float v = AGG[i] + HW[i] * d * d + ((f < biasN) ? bias[f] : 0.0f);
  if (flags & 2) v = fmaxf(v, 0.0f);
  if (flags & 4) v *= dropmask((uint32_t)i, seed);
  if (flags & 8) outB[i] = f2bf(v);
  else           outF[i] = v;
}

// P[rows[e]] += vals[e]*G[cols[e]]; one thread per (nnz, 4 features), 10 quads
__global__ void spmm_coo(const int* __restrict__ rows, const int* __restrict__ cols,
                         const float* __restrict__ vals, const float* __restrict__ G,
                         float* __restrict__ P, int nnz) {
  long long i = (long long)blockIdx.x * blockDim.x + threadIdx.x;
  if (i >= (long long)nnz * (OUT_F / 4)) return;
  const int e = (int)(i / (OUT_F / 4));
  const int q = (int)(i % (OUT_F / 4)) * 4;
  const float v = vals[e];
  const float4 g = *(const float4*)(G + (size_t)cols[e] * OUT_P + q);
  float* p = P + (size_t)rows[e] * OUT_P + q;
  atomicAdd(p + 0, g.x * v);
  atomicAdd(p + 1, g.y * v);
  atomicAdd(p + 2, g.z * v);
  atomicAdd(p + 3, g.w * v);
}

// out = log_softmax(p + p*c) over 40 columns, one thread per row
__global__ void logsoftmax_out(const float* __restrict__ P, const float* __restrict__ Cc,
                               float* __restrict__ out, int N) {
  int i = blockIdx.x * blockDim.x + threadIdx.x;
  if (i >= N) return;
  float v[OUT_F];
  float mx = -3.402823466e38f;
#pragma unroll
  for (int j = 0; j < OUT_F; ++j) {
    float p = P[(size_t)i * OUT_P + j];
    float t = p + p * Cc[(size_t)i * OUT_P + j];
    v[j] = t;
    mx = fmaxf(mx, t);
  }
  float s = 0.0f;
#pragma unroll
  for (int j = 0; j < OUT_F; ++j) s += expf(v[j] - mx);
  float ls = logf(s);
#pragma unroll
  for (int j = 0; j < OUT_F; ++j) out[(size_t)i * OUT_F + j] = v[j] - mx - ls;
}

// ---------------------------------------------------------------------------

static inline int cdiv(long long a, long long b) { return (int)((a + b - 1) / b); }

extern "C" void kernel_launch(void* const* d_in, const int* in_sizes, int n_in,
                              void* d_out, int out_size, void* d_ws, size_t ws_size,
                              hipStream_t stream) {
  const float* x     = (const float*)d_in[0];
  const float* omega = (const float*)d_in[1];
  const int*   ei    = (const int*)d_in[2];
  const int*   prow  = (const int*)d_in[3];
  const int*   pcol  = (const int*)d_in[4];
  const float* pval  = (const float*)d_in[5];
  const float* Wg1 = (const float*)d_in[6];  const float* bg1 = (const float*)d_in[7];
  const float* Wg2 = (const float*)d_in[8];  const float* bg2 = (const float*)d_in[9];
  const float* Wg3 = (const float*)d_in[10]; const float* bg3 = (const float*)d_in[11];
  const float* Wc1 = (const float*)d_in[12]; const float* bc1 = (const float*)d_in[13];
  const float* Wc2 = (const float*)d_in[14]; const float* bc2 = (const float*)d_in[15];
  const float* Wc3 = (const float*)d_in[16]; const float* bc3 = (const float*)d_in[17];
  float* out = (float*)d_out;

  const int N  = in_sizes[0] / IN_F;   // 50000 (multiple of 16)
  const int E  = in_sizes[2] / 2;      // 800000
  const int PN = in_sizes[3];          // 100000
  const int* src  = ei;
  const int* dstp = ei + E;

  // workspace carve-out (float units; ~160 MB total)
  float* ws = (float*)d_ws;
  size_t o = 0;
  float*  dis = ws + o; o += ((size_t)N + 63) & ~(size_t)63;
  float*  t0  = ws + o; o += (size_t)N * HID_F;          // HW (f32)  [N,<=128]
  float*  t1  = ws + o; o += (size_t)N * HID_F;          // AGG (f32) [N,<=128]
  float*  g   = ws + o; o += (size_t)N * OUT_P;          // GCN out   [N,48]
  float*  p   = ws + o; o += (size_t)N * OUT_P;          // spmm out  [N,48]
  float*  cc  = ws + o; o += (size_t)N * OUT_P;          // comp out  [N,48]
  __bf16* abf = (__bf16*)(ws + o); o += (size_t)N * (IN_F / 2);   // bf16 act A [N,<=512]
  __bf16* bbf = (__bf16*)(ws + o); o += (size_t)N * (CHID_F / 2); // bf16 act B [N,<=256]
  __bf16* Wg1t = (__bf16*)(ws + o);
  __bf16* Wg2t = Wg1t + (size_t)HID_F * IN_F;
  __bf16* Wg3t = Wg2t + (size_t)HID_F * HID_F;
  __bf16* Wc1t = Wg3t + (size_t)OUT_P * HID_F;
  __bf16* Wc2t = Wc1t + (size_t)CHID_F * IN_F;
  __bf16* Wc3t = Wc2t + (size_t)CHID_F * CHID_F;

  // 1) weights -> transposed bf16
  wconv<<<cdiv((long long)HID_F * IN_F,    256), 256, 0, stream>>>(Wg1, Wg1t, IN_F,   HID_F,  HID_F);
  wconv<<<cdiv((long long)HID_F * HID_F,   256), 256, 0, stream>>>(Wg2, Wg2t, HID_F,  HID_F,  HID_F);
  wconv<<<cdiv((long long)OUT_P * HID_F,   256), 256, 0, stream>>>(Wg3, Wg3t, HID_F,  OUT_F,  OUT_P);
  wconv<<<cdiv((long long)CHID_F * IN_F,   256), 256, 0, stream>>>(Wc1, Wc1t, IN_F,   CHID_F, CHID_F);
  wconv<<<cdiv((long long)CHID_F * CHID_F, 256), 256, 0, stream>>>(Wc2, Wc2t, CHID_F, CHID_F, CHID_F);
  wconv<<<cdiv((long long)OUT_P * CHID_F,  256), 256, 0, stream>>>(Wc3, Wc3t, CHID_F, OUT_F,  OUT_P);

  // 2) dis = rsqrt(1 + in-degree)
  fill_one <<<cdiv(N, 256), 256, 0, stream>>>(dis, N);
  deg_edges<<<cdiv(E, 256), 256, 0, stream>>>(dstp, dis, E);
  to_rsqrt <<<cdiv(N, 256), 256, 0, stream>>>(dis, N);

  const int gblocks = cdiv(cdiv(N, 16), 4);  // 4 waves (16-row stripes) / block

  // 3) GCN layer 1
  cvtbf<<<cdiv((long long)N * IN_F / 4, 256), 256, 0, stream>>>(x, abf, (long long)N * IN_F / 4);
  gemm_bf16_wmma<4><<<gblocks, 128, 0, stream>>>(abf, IN_F, Wg1t, HID_F, nullptr, 0, t0, nullptr, N, 0, 0);
  hipMemsetAsync(t1, 0, (size_t)N * HID_F * sizeof(float), stream);
  edge_agg<<<cdiv((long long)E * (HID_F / 4), 256), 256, 0, stream>>>(src, dstp, dis, t0, t1, E, HID_F, HID_F / 4);
  gcn_finalize<<<cdiv((long long)N * HID_F, 256), 256, 0, stream>>>(t1, t0, dis, bg1, N, HID_F, HID_F, 2 | 4 | 8, 0x12345687u, nullptr, abf);

  // 4) GCN layer 2
  gemm_bf16_wmma<4><<<gblocks, 128, 0, stream>>>(abf, HID_F, Wg2t, HID_F, nullptr, 0, t0, nullptr, N, 0, 0);
  hipMemsetAsync(t1, 0, (size_t)N * HID_F * sizeof(float), stream);
  edge_agg<<<cdiv((long long)E * (HID_F / 4), 256), 256, 0, stream>>>(src, dstp, dis, t0, t1, E, HID_F, HID_F / 4);
  gcn_finalize<<<cdiv((long long)N * HID_F, 256), 256, 0, stream>>>(t1, t0, dis, bg2, N, HID_F, HID_F, 2 | 4 | 8, 0x89abcdefu, nullptr, abf);

  // 5) GCN layer 3 (padded to 48 cols, no relu/dropout, f32 out)
  gemm_bf16_wmma<3><<<gblocks, 128, 0, stream>>>(abf, HID_F, Wg3t, OUT_P, nullptr, 0, t0, nullptr, N, 0, 0);
  hipMemsetAsync(g, 0, (size_t)N * OUT_P * sizeof(float), stream);
  edge_agg<<<cdiv((long long)E * (OUT_P / 4), 256), 256, 0, stream>>>(src, dstp, dis, t0, g, E, OUT_P, OUT_P / 4);
  gcn_finalize<<<cdiv((long long)N * OUT_P, 256), 256, 0, stream>>>(g, t0, dis, bg3, N, OUT_P, OUT_F, 0, 0, g, nullptr);

  // 6) p = partition @ g (COO spmm)
  hipMemsetAsync(p, 0, (size_t)N * OUT_P * sizeof(float), stream);
  spmm_coo<<<cdiv((long long)PN * (OUT_F / 4), 256), 256, 0, stream>>>(prow, pcol, pval, g, p, PN);

  // 7) compensation MLP: bf16-in/bf16-out chained GEMMs
  cvtbf<<<cdiv((long long)N * IN_F / 4, 256), 256, 0, stream>>>(omega, abf, (long long)N * IN_F / 4);
  gemm_bf16_wmma<4><<<gblocks, 128, 0, stream>>>(abf, IN_F,   Wc1t, CHID_F, bc1, CHID_F, nullptr, bbf, N, 1 | 2 | 4 | 8, 0x51f15eedu);
  gemm_bf16_wmma<4><<<gblocks, 128, 0, stream>>>(bbf, CHID_F, Wc2t, CHID_F, bc2, CHID_F, nullptr, abf, N, 1 | 2 | 4 | 8, 0xc0ffee11u);
  gemm_bf16_wmma<3><<<gblocks, 128, 0, stream>>>(abf, CHID_F, Wc3t, OUT_P,  bc3, OUT_F,  cc, nullptr, N, 1, 0);

  // 8) out = log_softmax(p * (1 + c))
  logsoftmax_out<<<cdiv(N, 128), 128, 0, stream>>>(p, cc, out, N);
}
